// AdMSoftmaxLoss_77524159693169
// MI455X (gfx1250) — matrix operand order repaired and checked by reading, not compiled, not run on hardware
//
#include <hip/hip_runtime.h>
#include <hip/hip_bf16.h>
#include <math.h>

// Problem constants (match reference)
#define PN 16384
#define PD 256
#define PC 10000
#define PS 30.0f
#define PM 0.4f
#define CTILES (PC / 16)   // 625 class tiles of 16
#define NPART 2            // class-dimension split for occupancy

typedef __attribute__((ext_vector_type(16))) __bf16 v16bf;
typedef __attribute__((ext_vector_type(8)))  __bf16 v8bf;
typedef __attribute__((ext_vector_type(8)))  float  v8f;

// ---------------------------------------------------------------------------
// Kernel 1: split W into bf16 hi/lo pair (w ~= hi + lo), elementwise.
// ---------------------------------------------------------------------------
__global__ void k_split_w(const float* __restrict__ w,
                          __bf16* __restrict__ hi, __bf16* __restrict__ lo,
                          int n) {
  for (int i = blockIdx.x * blockDim.x + threadIdx.x; i < n;
       i += gridDim.x * blockDim.x) {
    float v = w[i];
    __bf16 h = (__bf16)v;
    hi[i] = h;
    lo[i] = (__bf16)(v - (float)h);
  }
}

// ---------------------------------------------------------------------------
// Kernel 2: L2-normalize each row of x, then bf16 hi/lo split.
// One wave32 per row; 8 floats per lane; shfl-xor reduction for sum-of-squares.
// ---------------------------------------------------------------------------
__global__ void k_norm_split_x(const float* __restrict__ x,
                               __bf16* __restrict__ hi,
                               __bf16* __restrict__ lo) {
  const int wave = threadIdx.x >> 5;
  const int lane = threadIdx.x & 31;
  const int row  = blockIdx.x * 8 + wave;
  const float4* xr = (const float4*)(x + (size_t)row * PD);
  float4 v0 = xr[lane];
  float4 v1 = xr[lane + 32];
  float ss = v0.x * v0.x + v0.y * v0.y + v0.z * v0.z + v0.w * v0.w +
             v1.x * v1.x + v1.y * v1.y + v1.z * v1.z + v1.w * v1.w;
#pragma unroll
  for (int m = 16; m >= 1; m >>= 1) ss += __shfl_xor(ss, m, 32);
  const float inv = 1.0f / sqrtf(ss);

  float e[8] = {v0.x, v0.y, v0.z, v0.w, v1.x, v1.y, v1.z, v1.w};
  size_t base = (size_t)row * PD;
#pragma unroll
  for (int j = 0; j < 8; ++j) {
    float nv = e[j] * inv;
    __bf16 h = (__bf16)nv;
    size_t idx = base + (size_t)(j < 4 ? lane * 4 + j : 128 + lane * 4 + (j - 4));
    hi[idx] = h;
    lo[idx] = (__bf16)(nv - (float)h);
  }
}

// ---------------------------------------------------------------------------
// Kernel 3: exact fp32 target logit  tgt[i] = dot(x[i], W[label_i]) / ||x[i]||
// ---------------------------------------------------------------------------
__global__ void k_target(const float* __restrict__ x,
                         const float* __restrict__ w,
                         const int* __restrict__ labels,
                         float* __restrict__ tgt) {
  const int wave = threadIdx.x >> 5;
  const int lane = threadIdx.x & 31;
  const int row  = blockIdx.x * 8 + wave;
  const int cls  = labels[row];
  const float4* xr = (const float4*)(x + (size_t)row * PD);
  const float4* wr = (const float4*)(w + (size_t)cls * PD);
  float4 a0 = xr[lane], a1 = xr[lane + 32];
  float4 b0 = wr[lane], b1 = wr[lane + 32];
  float ss = a0.x * a0.x + a0.y * a0.y + a0.z * a0.z + a0.w * a0.w +
             a1.x * a1.x + a1.y * a1.y + a1.z * a1.z + a1.w * a1.w;
  float dp = a0.x * b0.x + a0.y * b0.y + a0.z * b0.z + a0.w * b0.w +
             a1.x * b1.x + a1.y * b1.y + a1.z * b1.z + a1.w * b1.w;
#pragma unroll
  for (int m = 16; m >= 1; m >>= 1) {
    ss += __shfl_xor(ss, m, 32);
    dp += __shfl_xor(dp, m, 32);
  }
  if (lane == 0) tgt[row] = dp / sqrtf(ss);
}

// ---------------------------------------------------------------------------
// Kernel 4: fused GEMM + exp-row-sum with LDS-staged B tiles.
// Block = 256 threads (8 waves) owns 128 rows; each wave owns 16 rows with
// register-resident A fragments (hi+lo, K=256). Class tiles (16 classes each)
// are staged into double-buffered LDS once per block and consumed by all 8
// waves (8x cut in L2 traffic for W). blockIdx.y splits the class dim 2-ways
// into deterministic per-row partial sums. 3 bf16 WMMAs per K-step recover
// ~fp32 precision (hi*hi + hi*lo + lo*hi).
// ---------------------------------------------------------------------------
__global__ void __launch_bounds__(256, 1)
k_gemm_expsum(const __bf16* __restrict__ xh, const __bf16* __restrict__ xl,
              const __bf16* __restrict__ wh, const __bf16* __restrict__ wl,
              float* __restrict__ rowSumPart) {
  constexpr int LROW = 264;  // 256 + 8 pad elements (528B row stride)
  __shared__ alignas(16) __bf16 ldsH[2][16 * LROW];
  __shared__ alignas(16) __bf16 ldsL[2][16 * LROW];

  const int tid  = threadIdx.x;
  const int lane = tid & 31;
  const int wave = tid >> 5;
  const int row0 = blockIdx.x * 128 + wave * 16;  // 16 rows owned by this wave
  const int part = blockIdx.y;                    // class tiles part, part+2, ...
  const int m    = lane & 15;        // row within A tile / class within B tile
  const int kA   = (lane >> 4) * 8;  // A half-lane K chunk offset
  const int kB   = (lane >> 4) * 8;  // B half-lane K chunk offset

  // Register-resident A fragments (16-bit A layout: two 8-elem K chunks/lane)
  v16bf ah[8], al[8];
  {
    const __bf16* ph = xh + (size_t)(row0 + m) * PD + kA;
    const __bf16* pl = xl + (size_t)(row0 + m) * PD + kA;
#pragma unroll
    for (int ks = 0; ks < 8; ++ks) {
      v8bf h0 = *(const v8bf*)(ph + ks * 32);
      v8bf h1 = *(const v8bf*)(ph + ks * 32 + 16);
      v8bf l0 = *(const v8bf*)(pl + ks * 32);
      v8bf l1 = *(const v8bf*)(pl + ks * 32 + 16);
#pragma unroll
      for (int j = 0; j < 8; ++j) {
        ah[ks][j] = h0[j]; ah[ks][8 + j] = h1[j];
        al[ks][j] = l0[j]; al[ks][8 + j] = l1[j];
      }
    }
  }

  const int nt = (CTILES - part + 1) / NPART;  // 313 / 312 tiles

  // B-tile staging registers: 4 x 16B chunks per thread (hi: j=0,1; lo: j=2,3)
  uint4 reg[4];
  auto ld_regs = [&](int ct) {
    const size_t cls0 = (size_t)ct * 16;
#pragma unroll
    for (int j = 0; j < 4; ++j) {
      int ch = j * 256 + tid;        // chunk id within matrix half
      int cc = (ch & 511) >> 5;      // class 0..15
      int k8 = ch & 31;              // 8-element K chunk
      const __bf16* src = (j < 2 ? wh : wl) + (cls0 + cc) * PD + k8 * 8;
      reg[j] = *(const uint4*)src;
    }
  };

  v8f se = {};  // per-lane running exp sums (VGPR r -> row r / r+8)
  int buf = 0;
  ld_regs(part);

  for (int i = 0; i < nt; ++i) {
    // Stage current B tile into LDS
#pragma unroll
    for (int j = 0; j < 4; ++j) {
      int ch = j * 256 + tid;
      int cc = (ch & 511) >> 5;
      int k8 = ch & 31;
      __bf16* dst = (j < 2 ? ldsH[buf] : ldsL[buf]) + cc * LROW + k8 * 8;
      *(uint4*)dst = reg[j];
    }
    __syncthreads();
    if (i + 1 < nt) ld_regs(part + (i + 1) * NPART);  // prefetch next tile

    // Compute from LDS: 8 K-steps x 3 WMMAs
    const __bf16* lbh = ldsH[buf] + m * LROW + kB;
    const __bf16* lbl = ldsL[buf] + m * LROW + kB;
    v8f acc = {};
#pragma unroll
    for (int ks = 0; ks < 8; ++ks) {
      v8bf h0 = *(const v8bf*)(lbh + ks * 32);
      v8bf h1 = *(const v8bf*)(lbh + ks * 32 + 16);
      v8bf l0 = *(const v8bf*)(lbl + ks * 32);
      v8bf l1 = *(const v8bf*)(lbl + ks * 32 + 16);
      v16bf fbh, fbl;
#pragma unroll
      for (int j = 0; j < 8; ++j) {
        fbh[j] = h0[j]; fbh[8 + j] = h1[j];
        fbl[j] = l0[j]; fbl[8 + j] = l1[j];
      }
      acc = __builtin_amdgcn_wmma_f32_16x16x32_bf16(false, ah[ks], false, fbh,
                                                    (short)0, acc, false, false);
      acc = __builtin_amdgcn_wmma_f32_16x16x32_bf16(false, ah[ks], false, fbl,
                                                    (short)0, acc, false, false);
      acc = __builtin_amdgcn_wmma_f32_16x16x32_bf16(false, al[ks], false, fbh,
                                                    (short)0, acc, false, false);
    }
#pragma unroll
    for (int r = 0; r < 8; ++r) se[r] += __expf(PS * acc[r]);
    buf ^= 1;
    // NOTE: single barrier per iter is safe: sync(i+1) happens after every
    // wave finished compute(i), and store(i+1) targets the other buffer.
  }

  // Reduce over the 16 classes held across each 16-lane half.
#pragma unroll
  for (int r = 0; r < 8; ++r) {
    float s = se[r];
    s += __shfl_xor(s, 1, 32);
    s += __shfl_xor(s, 2, 32);
    s += __shfl_xor(s, 4, 32);
    s += __shfl_xor(s, 8, 32);
    se[r] = s;
  }
  float* dst = rowSumPart + (size_t)part * PN + row0;
  if (lane == 0) {
#pragma unroll
    for (int r = 0; r < 8; ++r) dst[r] = se[r];
  }
  if (lane == 16) {
#pragma unroll
    for (int r = 0; r < 8; ++r) dst[8 + r] = se[r];
  }
}

// ---------------------------------------------------------------------------
// Kernel 5: per-row loss, block-reduced partials (deterministic, no atomics).
// ---------------------------------------------------------------------------
__global__ void k_loss_partial(const float* __restrict__ rowSumPart,
                               const float* __restrict__ tgt,
                               float* __restrict__ partial) {
  __shared__ float sm[256];
  const int i = blockIdx.x * 256 + threadIdx.x;
  float t    = tgt[i];
  float num  = PS * (t - PM);
  float sumE = rowSumPart[i] + rowSumPart[PN + i];
  float excl = sumE - __expf(PS * t);
  float den  = __expf(num) + excl;
  float L    = num - __logf(den);
  sm[threadIdx.x] = L;
  __syncthreads();
#pragma unroll
  for (int s = 128; s > 0; s >>= 1) {
    if (threadIdx.x < s) sm[threadIdx.x] += sm[threadIdx.x + s];
    __syncthreads();
  }
  if (threadIdx.x == 0) partial[blockIdx.x] = sm[0];
}

__global__ void k_loss_final(const float* __restrict__ partial,
                             float* __restrict__ out) {
  __shared__ float sm[64];
  sm[threadIdx.x] = partial[threadIdx.x];
  __syncthreads();
#pragma unroll
  for (int s = 32; s > 0; s >>= 1) {
    if (threadIdx.x < s) sm[threadIdx.x] += sm[threadIdx.x + s];
    __syncthreads();
  }
  if (threadIdx.x == 0) out[0] = -sm[0] / (float)PN;
}

// ---------------------------------------------------------------------------
// Launch
// ---------------------------------------------------------------------------
extern "C" void kernel_launch(void* const* d_in, const int* in_sizes, int n_in,
                              void* d_out, int out_size, void* d_ws, size_t ws_size,
                              hipStream_t stream) {
  const float* x      = (const float*)d_in[0];
  const int*   labels = (const int*)d_in[1];   // harness delivers integer inputs as int32
  const float* W      = (const float*)d_in[2];
  float*       out    = (float*)d_out;

  char* ws = (char*)d_ws;
  // Workspace layout (all 256B-aligned by construction):
  __bf16* xnh     = (__bf16*)(ws);               // 16384*256*2 = 8,388,608
  __bf16* xnl     = (__bf16*)(ws + 8388608);     // 8,388,608
  __bf16* whi     = (__bf16*)(ws + 16777216);    // 10000*256*2 = 5,120,000
  __bf16* wlo     = (__bf16*)(ws + 21897216);    // 5,120,000
  float*  rowSumP = (float*)(ws + 27017216);     // 2*16384*4 = 131,072
  float*  tgt     = (float*)(ws + 27148288);     // 65,536
  float*  part    = (float*)(ws + 27213824);     // 64*4

  k_split_w<<<4096, 256, 0, stream>>>(W, whi, wlo, PC * PD);
  k_norm_split_x<<<PN / 8, 256, 0, stream>>>(x, xnh, xnl);
  k_target<<<PN / 8, 256, 0, stream>>>(x, W, labels, tgt);
  k_gemm_expsum<<<dim3(PN / 128, NPART), 256, 0, stream>>>(xnh, xnl, whi, wlo, rowSumP);
  k_loss_partial<<<PN / 256, 256, 0, stream>>>(rowSumP, tgt, part);
  k_loss_final<<<1, 64, 0, stream>>>(part, out);
}
